// DownsampleBlock_420906795541
// MI455X (gfx1250) — matrix operand order repaired and checked by reading, not compiled, not run on hardware
//
#include <hip/hip_runtime.h>
#include <hip/hip_bf16.h>

typedef _Float16 v16h __attribute__((ext_vector_type(16)));
typedef _Float16 v8h  __attribute__((ext_vector_type(8)));
typedef _Float16 v4h  __attribute__((ext_vector_type(4)));
typedef float    v8f  __attribute__((ext_vector_type(8)));
typedef float    v4f  __attribute__((ext_vector_type(4)));

#define CIN    256
#define COUT   512
#define KNN    16
#define APITCH 264   // 256 + 8 halves pad -> 528B row pitch, 16B aligned

__global__ __launch_bounds__(256)
void downsample_fused_kernel(const float* __restrict__ x,
                             const int*   __restrict__ idx,
                             const float* __restrict__ gamma,
                             const float* __restrict__ beta,
                             const float* __restrict__ W,
                             const float* __restrict__ bias,
                             float*       __restrict__ out,
                             int Mtot)
{
    // Only the double-buffered A tile lives in LDS (~17 KB total).
    __shared__ __align__(16) _Float16 sA[2][KNN * APITCH];

    const int tid    = threadIdx.x;
    const int lane   = tid & 31;
    const int wave   = tid >> 5;
    const int c16    = lane & 15;      // lane within half-wave
    const int half   = lane >> 4;      // 0 or 1
    const int koff_a = half << 4;      // A fragment K offset: 0 or 16 (contiguous)
    const int koff_b = half << 4;      // B fragment K offset: 0 or 16 (contiguous)

    // ---- one-time: B fragments (W^T, f16) resident in VGPRs for the kernel ----
    // Fragment (t,c): lane col n=(wave*4+t)*16+c16, K = c*32 + koff_b + 0..15.
    v16h bfrag[4][8];
#pragma unroll
    for (int t = 0; t < 4; ++t) {
        const int n = (wave * 4 + t) * 16 + c16;
#pragma unroll
        for (int c = 0; c < 8; ++c) {
            const float* wp = W + (size_t)(c * 32 + koff_b) * COUT + n;
            v16h b;
#pragma unroll
            for (int i = 0; i < 16; ++i)
                b[i] = (_Float16)wp[(size_t)i * COUT];   // 64B coalesced per half-wave
            bfrag[t][c] = b;
        }
    }

    // ---- one-time: per-lane gamma/beta channels (loop-invariant) ----
    // This lane owns channels ch = j*64 + c16*4, j = 0..3.
    v4f ga[4], be[4];
#pragma unroll
    for (int j = 0; j < 4; ++j) {
        const int ch = j * 64 + c16 * 4;
        ga[j] = *(const v4f*)(gamma + ch);
        be[j] = *(const v4f*)(beta + ch);
    }

    // wave w owns neighbor rows 2w (lanes 0-15) and 2w+1 (lanes 16-31)
    const int r = wave * 2 + half;

    // ---- prefetch gather for first mi ----
    int mi = blockIdx.x;
    v4f g[4];
    if (mi < Mtot) {
        const int nb = idx[mi * KNN + r];
        const float* xrow = x + (size_t)nb * CIN;
#pragma unroll
        for (int j = 0; j < 4; ++j)
            g[j] = *(const v4f*)(xrow + j * 64 + c16 * 4);
    }

    int parity = 0;
    for (; mi < Mtot; mi += gridDim.x) {
        // ---------- phase 1: LayerNorm on prefetched rows -> f16 A[parity] ----------
        float s = 0.f, s2 = 0.f;
#pragma unroll
        for (int j = 0; j < 4; ++j)
#pragma unroll
            for (int i = 0; i < 4; ++i) { float v = g[j][i]; s += v; s2 += v * v; }
#pragma unroll
        for (int m = 1; m <= 8; m <<= 1) {   // reduce over 16-lane half-wave
            s  += __shfl_xor(s,  m, 32);
            s2 += __shfl_xor(s2, m, 32);
        }
        const float mean = s * (1.0f / CIN);
        const float var  = s2 * (1.0f / CIN) - mean * mean;
        const float rstd = rsqrtf(var + 1e-5f);

        _Float16* arow = &sA[parity][r * APITCH];
#pragma unroll
        for (int j = 0; j < 4; ++j) {
            v4h h4;
#pragma unroll
            for (int i = 0; i < 4; ++i)
                h4[i] = (_Float16)(((g[j][i] - mean) * rstd) * ga[j][i] + be[j][i]);
            *(v4h*)(arow + j * 64 + c16 * 4) = h4;       // ds_store_b64
        }

        __syncthreads();   // A[parity] complete; prev iter's reads of A[parity^1] done

        // ---------- phase 2a: issue gather for next mi (hides under WMMA) ----------
        const int mnext = mi + gridDim.x;
        if (mnext < Mtot) {
            const int nb = idx[mnext * KNN + r];
            const float* xrow = x + (size_t)nb * CIN;
#pragma unroll
            for (int j = 0; j < 4; ++j)
                g[j] = *(const v4f*)(xrow + j * 64 + c16 * 4);
        }

        // ---------- phase 2b: WMMA [16x256]x[256x512], B from registers ----------
        v8f acc[4];
#pragma unroll
        for (int t = 0; t < 4; ++t) acc[t] = (v8f){};

        const _Float16* Ab = sA[parity];
#pragma unroll
        for (int c = 0; c < 8; ++c) {
            // A fragment: lane row = c16; halves 0-15 = K c*32+koff_a+0..15
            // (contiguous) -> two b128 at +0 and +16B.
            const _Float16* ap = &Ab[c16 * APITCH + c * 32 + koff_a];
            v8h lo = *(const v8h*)ap;
            v8h hi = *(const v8h*)(ap + 8);
            v16h a;
#pragma unroll
            for (int i = 0; i < 8; ++i) { a[i] = lo[i]; a[i + 8] = hi[i]; }
#pragma unroll
            for (int t = 0; t < 4; ++t)
                acc[t] = __builtin_amdgcn_wmma_f32_16x16x32_f16(
                    false, a, false, bfrag[t][c], (short)0, acc[t], false, false);
        }

        // ---------- phase 3: max over 16 neighbor rows + bias + store ----------
        // C/D layout: VGPR j of lane = row M=j+8*half, col N=c16.
#pragma unroll
        for (int t = 0; t < 4; ++t) {
            float mx = acc[t][0];
#pragma unroll
            for (int i = 1; i < 8; ++i) mx = fmaxf(mx, acc[t][i]);
            mx = fmaxf(mx, __shfl_xor(mx, 16, 32));      // rows 0-7 vs 8-15
            const int n = (wave * 4 + t) * 16 + c16;
            if (half == (t & 1))                          // split stores across halves
                out[(size_t)mi * COUT + n] = mx + bias[n];
        }

        parity ^= 1;   // no trailing barrier needed: next write hits other buffer
    }
}

extern "C" void kernel_launch(void* const* d_in, const int* in_sizes, int n_in,
                              void* d_out, int out_size, void* d_ws, size_t ws_size,
                              hipStream_t stream) {
    const float* x     = (const float*)d_in[0];
    const int*   idx   = (const int*)  d_in[1];
    const float* gamma = (const float*)d_in[2];
    const float* beta  = (const float*)d_in[3];
    const float* W     = (const float*)d_in[4];
    const float* bias  = (const float*)d_in[5];
    float*       out   = (float*)d_out;

    const int Mtot = in_sizes[1] / KNN;      // idx has M*K elements
    int grid = Mtot < 512 ? Mtot : 512;
    downsample_fused_kernel<<<dim3(grid), dim3(256), 0, stream>>>(
        x, idx, gamma, beta, W, bias, out, Mtot);
}
